// DNET_27754078667079
// MI455X (gfx1250) — matrix operand order repaired and checked by reading, not compiled, not run on hardware
//
#include <hip/hip_runtime.h>
#include <math.h>

// ---------------------------------------------------------------------------
// CDNA5 / gfx1250 fused implementation of the iterative registration network.
// All channel-MLPs over the N*N pair grid run on v_wmma_f32_16x16x32_bf16.
// ---------------------------------------------------------------------------

typedef __attribute__((ext_vector_type(16))) __bf16 v16bf;
typedef __attribute__((ext_vector_type(8)))  float  v8f;

#define NPT 512
#define NB  2

// -------------------- workspace layout (bytes) -----------------------------
#define WOFF_BF16  0            // packed bf16 weights (~56 KB)
#define WOFF_BIAS  (64*1024)    // folded biases + sim_mat2 L2 f32 (~2.2 KB)
#define WOFF_SRC   (96*1024)    // current src points  B*3*N f32
#define WOFF_RACC  (112*1024)   // R acc (B*9) then t acc (B*3)
#define WOFF_WMKEY (128*1024)   // B*16*N u32 order-keys for float max
#define WOFF_WVEC  (192*1024)   // sigmoid(weights) B*N f32
#define WOFF_WNORM (200*1024)   // median-masked normalized weights
#define WOFF_CORR  (216*1024)   // argmax correspondence B*N int
#define WOFF_SIM   (256*1024)   // sim map B*N*N f32 (2 MB)

// -------------------- packed-weight offsets (bf16 elems) -------------------
#define OFF_SS1(i)  ((i)*7168)          // sim_spatial L1 [32][224]
#define OFF_SS2(i)  (14336 + (i)*1024)  // sim_spatial L2 [32][32]
#define OFF_ED1     16384               // embedding_dis L1 [32][64]
#define OFF_ED2     18432               // embedding_dis L2 [32pad][32]
#define OFF_SO1(i)  (19456 + (i)*1024)  // sim_origin L1 [32][32pad]
#define OFF_SO2(i)  (21504 + (i)*1024)  // sim_origin L2 [32][32]
#define OFF_FU      23552               // fuse [32][32]
#define OFF_SM1(i)  (24576 + (i)*1024)  // sim_mat L1 [32][32]
#define OFF_SM2(i)  (26624 + (i)*512)   // sim_mat L2 [16][32]
#define OFF_SMM1(i) (27648 + (i)*512)   // sim_mat2 L1 [16][32pad]

// -------------------- bias offsets (f32 elems) -----------------------------
#define B_SS1(i)  ((i)*32)
#define B_SS2(i)  (64 + (i)*32)
#define B_ED1     128
#define B_ED2     160
#define B_SO1(i)  (192 + (i)*32)
#define B_SO2(i)  (256 + (i)*32)
#define B_FU      320
#define B_SM1(i)  (352 + (i)*32)
#define B_SM2(i)  (416 + (i)*16)
#define B_SMM1(i) (448 + (i)*16)
#define B_SM2W(i) (480 + (i)*16)        // sim_mat2 L2 weights (f32, VALU dot)
#define B_SM2B(i) (512 + (i))

// -------------------- d_in indices (jax pytree, dict keys sorted) ----------
#define I_SRC 0
#define I_TGT 1
#define I_SE  2
#define I_TE  3
#define I_SP  6
#define I_TP  7
#define I_SOP 8
#define I_TOP 9
#define P_ED    11  // W,beta,g, W2,bias
#define P_FU    16  // W,bias
#define P_OD    18  // W,beta,g, W2,bias
#define P_SM(i)  (23 + 5*(i))
#define P_SM2(i) (33 + 5*(i))
#define P_SO(i)  (43 + 5*(i))
#define P_SS(i)  (53 + 5*(i))
#define P_WF(i)  (63 + 5*(i))

struct PtrPack { const float* in[80]; };

// ---------------------------------------------------------------------------
__device__ __forceinline__ unsigned short f2bf(float f) {
  unsigned u = __float_as_uint(f);
  u += 0x7FFFu + ((u >> 16) & 1u);           // RNE
  return (unsigned short)(u >> 16);
}
__device__ __forceinline__ unsigned fkey(float f) {   // order-preserving key
  unsigned u = __float_as_uint(f);
  return (u & 0x80000000u) ? ~u : (u | 0x80000000u);
}
__device__ __forceinline__ float unfkey(unsigned k) {
  return __uint_as_float((k & 0x80000000u) ? (k ^ 0x80000000u) : ~k);
}
__device__ __forceinline__ v8f zero8() {
  v8f z;
  #pragma unroll
  for (int i = 0; i < 8; ++i) z[i] = 0.f;
  return z;
}
__device__ __forceinline__ v8f wmma_bf16(v16bf a, v16bf b, v8f c) {
  return __builtin_amdgcn_wmma_f32_16x16x32_bf16(false, a, false, b,
                                                 (short)0, c, false, false);
}
// A-fragment (16x32 bf16): lanes0-15 M=lane K0..15, lanes16-31 K16..31
__device__ __forceinline__ v16bf afrag(const __bf16* W, int stride, int otile,
                                       int kstep, int c, int hi) {
  v16bf a;
  int row = otile * 16 + c;
  #pragma unroll
  for (int j = 0; j < 8; ++j) {
    int k0 = kstep * 32 + 2 * j + (hi ? 16 : 0);
    a[2 * j]     = W[row * stride + k0];
    a[2 * j + 1] = W[row * stride + k0 + 1];
  }
  return a;
}
// B-fragment from LDS tile [32][stride17]: lane covers 16 consecutive K
__device__ __forceinline__ v16bf b_from_lds(volatile float* l, int c, int hi) {
  v16bf b;
  int kb = hi ? 16 : 0;
  #pragma unroll
  for (int i = 0; i < 16; ++i) b[i] = (__bf16)l[(kb + i) * 17 + c];
  return b;
}
__device__ __forceinline__ void stage32(volatile float* l, v8f a0, v8f a1,
                                        int c, int hi) {
  int r0 = hi ? 8 : 0;
  #pragma unroll
  for (int r = 0; r < 8; ++r) {
    l[(r0 + r) * 17 + c]      = a0[r];
    l[(16 + r0 + r) * 17 + c] = a1[r];
  }
}
__device__ __forceinline__ v8f bias_act(v8f a, const float* bias, int hi,
                                        bool relu) {
  #pragma unroll
  for (int r = 0; r < 8; ++r) {
    float v = a[r] + bias[r + (hi ? 8 : 0)];
    a[r] = relu ? fmaxf(v, 0.f) : v;
  }
  return a;
}

// ---------------------------------------------------------------------------
// prep: fold BN into weights, pack bf16 A-layout tables, init state
// ---------------------------------------------------------------------------
__device__ void pack_bn(unsigned short* dst, float* bias, const float* W,
                        const float* beta, const float* g, int O, int K,
                        int Kpad, int tid, int nthr) {
  const float s = rsqrtf(1.f + 1e-5f);
  for (int idx = tid; idx < O * Kpad; idx += nthr) {
    int o = idx / Kpad, k = idx % Kpad;
    float v = (k < K) ? W[o * K + k] * g[o] * s : 0.f;
    dst[idx] = f2bf(v);
  }
  for (int o = tid; o < O; o += nthr) bias[o] = beta[o];
}
__device__ void pack_conv(unsigned short* dst, float* bias, const float* W,
                          const float* bvec, int O, int Opad, int K, int Kpad,
                          int tid, int nthr) {
  for (int idx = tid; idx < Opad * Kpad; idx += nthr) {
    int o = idx / Kpad, k = idx % Kpad;
    float v = (o < O && k < K) ? W[o * K + k] : 0.f;
    dst[idx] = f2bf(v);
  }
  for (int o = tid; o < Opad; o += nthr) bias[o] = (o < O) ? bvec[o] : 0.f;
}

__global__ void prep_kernel(PtrPack P, char* ws) {
  unsigned short* WB = (unsigned short*)(ws + WOFF_BF16);
  float* BB = (float*)(ws + WOFF_BIAS);
  float* srcC = (float*)(ws + WOFF_SRC);
  float* RT = (float*)(ws + WOFF_RACC);
  const int t = threadIdx.x, nt = blockDim.x;

  for (int i = 0; i < 2; ++i) {
    pack_bn(WB + OFF_SS1(i), BB + B_SS1(i), P.in[P_SS(i)], P.in[P_SS(i) + 1],
            P.in[P_SS(i) + 2], 32, 193, 224, t, nt);
    pack_conv(WB + OFF_SS2(i), BB + B_SS2(i), P.in[P_SS(i) + 3],
              P.in[P_SS(i) + 4], 32, 32, 32, 32, t, nt);
    pack_bn(WB + OFF_SO1(i), BB + B_SO1(i), P.in[P_SO(i)], P.in[P_SO(i) + 1],
            P.in[P_SO(i) + 2], 32, 10, 32, t, nt);
    pack_conv(WB + OFF_SO2(i), BB + B_SO2(i), P.in[P_SO(i) + 3],
              P.in[P_SO(i) + 4], 32, 32, 32, 32, t, nt);
    pack_bn(WB + OFF_SM1(i), BB + B_SM1(i), P.in[P_SM(i)], P.in[P_SM(i) + 1],
            P.in[P_SM(i) + 2], 32, 32, 32, t, nt);
    pack_conv(WB + OFF_SM2(i), BB + B_SM2(i), P.in[P_SM(i) + 3],
              P.in[P_SM(i) + 4], 16, 16, 32, 32, t, nt);
    pack_bn(WB + OFF_SMM1(i), BB + B_SMM1(i), P.in[P_SM2(i)],
            P.in[P_SM2(i) + 1], P.in[P_SM2(i) + 2], 16, 16, 32, t, nt);
    for (int k = t; k < 16; k += nt) BB[B_SM2W(i) + k] = P.in[P_SM2(i) + 3][k];
    if (t == 0) BB[B_SM2B(i)] = P.in[P_SM2(i) + 4][0];
  }
  pack_bn(WB + OFF_ED1, BB + B_ED1, P.in[P_ED], P.in[P_ED + 1],
          P.in[P_ED + 2], 32, 64, 64, t, nt);
  pack_conv(WB + OFF_ED2, BB + B_ED2, P.in[P_ED + 3], P.in[P_ED + 4],
            29, 32, 32, 32, t, nt);
  pack_conv(WB + OFF_FU, BB + B_FU, P.in[P_FU], P.in[P_FU + 1],
            32, 32, 32, 32, t, nt);

  for (int i = t; i < NB * 3 * NPT; i += nt) srcC[i] = P.in[I_SRC][i];
  for (int i = t; i < 24; i += nt) {
    float v = 0.f;
    if (i < 18) { int loc = i % 9; v = (loc == 0 || loc == 4 || loc == 8) ? 1.f : 0.f; }
    RT[i] = v;
  }
}

__global__ void zero_wmax(char* ws) {
  unsigned* k = (unsigned*)(ws + WOFF_WMKEY);
  int i = blockIdx.x * 256 + threadIdx.x;
  if (i < NB * 16 * NPT) k[i] = 0u;
}

// ---------------------------------------------------------------------------
// pair kernel: one wave per (b, src row n, 16 tgt cols). Full fused MLP chain.
// ---------------------------------------------------------------------------
__global__ __launch_bounds__(256) void pair_kernel(PtrPack P, char* ws,
                                                   int iter) {
  __shared__ float ldsAll[8][17 * 32];
  const int tid = threadIdx.x;
  const int wave = tid >> 5, lane = tid & 31;
  const int c = lane & 15, hi = lane >> 4;
  const int group = blockIdx.x * 8 + wave;
  const int b = group >> 14;
  const int n = (group >> 5) & 511;
  const int m0 = (group & 31) << 4;
  const int m = m0 + c;
  volatile float* lsm = (volatile float*)&ldsAll[wave][0];

  const __bf16* WB = (const __bf16*)(ws + WOFF_BF16);
  const float* BB = (const float*)(ws + WOFF_BIAS);
  const float* srcC = (const float*)(ws + WOFF_SRC);
  float* simbuf = (float*)(ws + WOFF_SIM);
  unsigned* wmkey = (unsigned*)(ws + WOFF_WMKEY);

  const float* se = P.in[I_SE];  const float* te = P.in[I_TE];
  const float* sp = P.in[I_SP];  const float* tp = P.in[I_TP];
  const float* sop = P.in[I_SOP]; const float* topp = P.in[I_TOP];
  const float* tgt = P.in[I_TGT];

  __builtin_prefetch(WB + OFF_SS1(iter), 0, 1);   // global_prefetch

  // per-lane geometry
  float src_n[3], tgt_m[3];
  #pragma unroll
  for (int k = 0; k < 3; ++k) {
    src_n[k] = srcC[(b * 3 + k) * NPT + n];
    tgt_m[k] = tgt[(b * 3 + k) * NPT + m];
  }
  float d2o = 0.f;
  #pragma unroll
  for (int k = 0; k < 3; ++k) { float d = src_n[k] - tgt_m[k]; d2o += d * d; }
  float dist_o = sqrtf(d2o), inv_do = 1.f / (dist_o + 1e-8f);
  float d2s = 0.f;
  for (int k = 0; k < 64; ++k) {
    float s = se[(b * 64 + k) * NPT + n], t = te[(b * 64 + k) * NPT + m];
    float d = s - t; d2s += d * d;
  }
  float dist_s = sqrtf(d2s), inv_ds = 1.f / (dist_s + 1e-8f);

  // ---- sim_spatial L1: 193(->224)->32 bn+relu -----------------------------
  v8f p0 = zero8(), p1 = zero8();
  {
    const __bf16* W = WB + OFF_SS1(iter);
    #pragma unroll
    for (int ks = 0; ks < 7; ++ks) {
      v16bf bx;
      int kb = ks * 32 + (hi ? 16 : 0);
      #pragma unroll
      for (int i = 0; i < 16; ++i) {
        int kk = kb + i; float v;
        if (kk < 64)        v = se[(b * 64 + kk) * NPT + n];
        else if (kk < 128)  v = te[(b * 64 + kk - 64) * NPT + m];
        else if (kk < 192) {
          int k = kk - 128;
          v = (se[(b * 64 + k) * NPT + n] - te[(b * 64 + k) * NPT + m]) * inv_ds;
        } else if (kk == 192) v = dist_s;
        else v = 0.f;
        bx[i] = (__bf16)v;
      }
      p0 = wmma_bf16(afrag(W, 224, 0, ks, c, hi), bx, p0);
      p1 = wmma_bf16(afrag(W, 224, 1, ks, c, hi), bx, p1);
    }
  }
  p0 = bias_act(p0, BB + B_SS1(iter), hi, true);
  p1 = bias_act(p1, BB + B_SS1(iter) + 16, hi, true);
  // L2 32->32 conv+bias
  stage32(lsm, p0, p1, c, hi);
  {
    v16bf bx = b_from_lds(lsm, c, hi);
    v8f a0 = zero8(), a1 = zero8();
    a0 = wmma_bf16(afrag(WB + OFF_SS2(iter), 32, 0, 0, c, hi), bx, a0);
    a1 = wmma_bf16(afrag(WB + OFF_SS2(iter), 32, 1, 0, c, hi), bx, a1);
    p0 = bias_act(a0, BB + B_SS2(iter), hi, false);
    p1 = bias_act(a1, BB + B_SS2(iter) + 16, hi, false);
  }
  // ---- sim_origin: 10->32 bn+relu, 32->32 conv ----------------------------
  {
    v16bf bo;
    #pragma unroll
    for (int i = 0; i < 16; ++i) {
      float v = 0.f;
      if (!hi) {
        if (i < 3)        v = src_n[i];
        else if (i < 6)   v = tgt_m[i - 3];
        else if (i < 9)   v = (src_n[i - 6] - tgt_m[i - 6]) * inv_do;
        else if (i == 9)  v = dist_o;
      }
      bo[i] = (__bf16)v;
    }
    v8f q0 = zero8(), q1 = zero8();
    q0 = wmma_bf16(afrag(WB + OFF_SO1(iter), 32, 0, 0, c, hi), bo, q0);
    q1 = wmma_bf16(afrag(WB + OFF_SO1(iter), 32, 1, 0, c, hi), bo, q1);
    q0 = bias_act(q0, BB + B_SO1(iter), hi, true);
    q1 = bias_act(q1, BB + B_SO1(iter) + 16, hi, true);
    stage32(lsm, q0, q1, c, hi);
    v16bf bx = b_from_lds(lsm, c, hi);
    q0 = zero8(); q1 = zero8();
    q0 = wmma_bf16(afrag(WB + OFF_SO2(iter), 32, 0, 0, c, hi), bx, q0);
    q1 = wmma_bf16(afrag(WB + OFF_SO2(iter), 32, 1, 0, c, hi), bx, q1);
    q0 = bias_act(q0, BB + B_SO2(iter), hi, false);
    q1 = bias_act(q1, BB + B_SO2(iter) + 16, hi, false);
    #pragma unroll
    for (int r = 0; r < 8; ++r) { p0[r] *= q0[r]; p1[r] *= q1[r]; }  // sm
  }
  // ---- embedding_dis: 64->32 bn+relu, 32->29 conv -> spatial_cc -----------
  v8f e0 = zero8(), e1 = zero8();
  {
    #pragma unroll
    for (int ks = 0; ks < 2; ++ks) {
      v16bf be;
      int kb = ks * 32 + (hi ? 16 : 0);
      #pragma unroll
      for (int i = 0; i < 16; ++i) {
        int kk = kb + i;
        float v = (se[(b * 64 + kk) * NPT + n] - sp[b * 64 + kk]) -
                  (te[(b * 64 + kk) * NPT + m] - tp[b * 64 + kk]);
        be[i] = (__bf16)v;
      }
      e0 = wmma_bf16(afrag(WB + OFF_ED1, 64, 0, ks, c, hi), be, e0);
      e1 = wmma_bf16(afrag(WB + OFF_ED1, 64, 1, ks, c, hi), be, e1);
    }
    e0 = bias_act(e0, BB + B_ED1, hi, true);
    e1 = bias_act(e1, BB + B_ED1 + 16, hi, true);
    stage32(lsm, e0, e1, c, hi);
    v16bf bx = b_from_lds(lsm, c, hi);
    e0 = zero8(); e1 = zero8();
    e0 = wmma_bf16(afrag(WB + OFF_ED2, 32, 0, 0, c, hi), bx, e0);
    e1 = wmma_bf16(afrag(WB + OFF_ED2, 32, 1, 0, c, hi), bx, e1);
    e0 = bias_act(e0, BB + B_ED2, hi, false);
    e1 = bias_act(e1, BB + B_ED2 + 16, hi, false);   // rows 29..31 are zero
  }
  // ---- origin_dis 3->3->3 in VALU -> origin_cc ----------------------------
  float oc[3];
  {
    const float* W1 = P.in[P_OD];     const float* be1 = P.in[P_OD + 1];
    const float* g1 = P.in[P_OD + 2]; const float* W2 = P.in[P_OD + 3];
    const float* b2 = P.in[P_OD + 4];
    float st[3], h3[3];
    #pragma unroll
    for (int k = 0; k < 3; ++k)
      st[k] = (src_n[k] - sop[b * 3 + k]) - (tgt_m[k] - topp[b * 3 + k]);
    const float sc = rsqrtf(1.f + 1e-5f);
    #pragma unroll
    for (int o = 0; o < 3; ++o) {
      float a = W1[o * 3] * st[0] + W1[o * 3 + 1] * st[1] + W1[o * 3 + 2] * st[2];
      h3[o] = fmaxf(a * g1[o] * sc + be1[o], 0.f);
    }
    #pragma unroll
    for (int o = 0; o < 3; ++o)
      oc[o] = W2[o * 3] * h3[0] + W2[o * 3 + 1] * h3[1] + W2[o * 3 + 2] * h3[2] + b2[o];
  }
  // ---- fuse: cc = sigmoid(conv32(concat(spatial_cc29, origin_cc3))) -------
  stage32(lsm, e0, e1, c, hi);
  if (!hi) { lsm[29 * 17 + c] = oc[0]; lsm[30 * 17 + c] = oc[1]; lsm[31 * 17 + c] = oc[2]; }
  {
    v16bf bx = b_from_lds(lsm, c, hi);
    v8f f0 = zero8(), f1 = zero8();
    f0 = wmma_bf16(afrag(WB + OFF_FU, 32, 0, 0, c, hi), bx, f0);
    f1 = wmma_bf16(afrag(WB + OFF_FU, 32, 1, 0, c, hi), bx, f1);
    f0 = bias_act(f0, BB + B_FU, hi, false);
    f1 = bias_act(f1, BB + B_FU + 16, hi, false);
    #pragma unroll
    for (int r = 0; r < 8; ++r) {
      p0[r] *= 1.f / (1.f + __expf(-f0[r]));
      p1[r] *= 1.f / (1.f + __expf(-f1[r]));
    }
  }
  // ---- sim_mat: 32->32 bn+relu, 32->16 conv -> sm16 -----------------------
  v8f sm16 = zero8();
  {
    stage32(lsm, p0, p1, c, hi);
    v16bf bx = b_from_lds(lsm, c, hi);
    v8f g0 = zero8(), g1 = zero8();
    g0 = wmma_bf16(afrag(WB + OFF_SM1(iter), 32, 0, 0, c, hi), bx, g0);
    g1 = wmma_bf16(afrag(WB + OFF_SM1(iter), 32, 1, 0, c, hi), bx, g1);
    g0 = bias_act(g0, BB + B_SM1(iter), hi, true);
    g1 = bias_act(g1, BB + B_SM1(iter) + 16, hi, true);
    stage32(lsm, g0, g1, c, hi);
    bx = b_from_lds(lsm, c, hi);
    sm16 = wmma_bf16(afrag(WB + OFF_SM2(iter), 32, 0, 0, c, hi), bx, sm16);
    sm16 = bias_act(sm16, BB + B_SM2(iter), hi, false);
  }
  // stage sm16 in ch0..15, zero 16..31
  #pragma unroll
  for (int r = 0; r < 8; ++r) {
    lsm[(r + (hi ? 8 : 0)) * 17 + c] = sm16[r];
    lsm[(16 + r + (hi ? 8 : 0)) * 17 + c] = 0.f;
  }
  // running row-max of sm over m (weights input), keyed u32 atomic max
  if (!hi) {
    float mx = -3.4e38f;
    #pragma unroll
    for (int j = 0; j < 16; ++j) mx = fmaxf(mx, lsm[lane * 17 + j]);
    atomicMax(&wmkey[(b * 16 + lane) * NPT + n], fkey(mx));
  }
  // ---- sim_mat2: 16->16 bn+relu (WMMA), 16->1 conv (VALU) -> sim ----------
  {
    v16bf bx = b_from_lds(lsm, c, hi);
    v8f h16 = zero8();
    h16 = wmma_bf16(afrag(WB + OFF_SMM1(iter), 32, 0, 0, c, hi), bx, h16);
    h16 = bias_act(h16, BB + B_SMM1(iter), hi, true);
    #pragma unroll
    for (int r = 0; r < 8; ++r) lsm[(r + (hi ? 8 : 0)) * 17 + c] = h16[r];
    if (!hi) {
      const float* w2 = BB + B_SM2W(iter);
      float accv = BB[B_SM2B(iter)];
      #pragma unroll
      for (int k = 0; k < 16; ++k) accv += w2[k] * lsm[k * 17 + c];
      accv = fminf(fmaxf(accv, -20.f), 20.f);
      simbuf[((size_t)(b * NPT + n)) * NPT + m] = accv;
    }
  }
}

// ---------------------------------------------------------------------------
// weights (weight_fc + sigmoid) and correspondence argmax, per (b,n)
// ---------------------------------------------------------------------------
__global__ void weights_kernel(PtrPack P, char* ws, int iter) {
  int idx = blockIdx.x * blockDim.x + threadIdx.x;
  if (idx >= NB * NPT) return;
  int b = idx >> 9, n = idx & 511;
  unsigned* wmkey = (unsigned*)(ws + WOFF_WMKEY);
  float x[16], h[16];
  #pragma unroll
  for (int o = 0; o < 16; ++o) x[o] = unfkey(wmkey[(b * 16 + o) * NPT + n]);
  const float* W1 = P.in[P_WF(iter)];     const float* be = P.in[P_WF(iter) + 1];
  const float* g  = P.in[P_WF(iter) + 2]; const float* W2 = P.in[P_WF(iter) + 3];
  const float* b2 = P.in[P_WF(iter) + 4];
  const float sc = rsqrtf(1.f + 1e-5f);
  #pragma unroll
  for (int o = 0; o < 16; ++o) {
    float a = 0.f;
    #pragma unroll
    for (int k = 0; k < 16; ++k) a += W1[o * 16 + k] * x[k];
    h[o] = fmaxf(a * g[o] * sc + be[o], 0.f);
  }
  float y = b2[0];
  #pragma unroll
  for (int k = 0; k < 16; ++k) y += W2[k] * h[k];
  ((float*)(ws + WOFF_WVEC))[idx] = 1.f / (1.f + __expf(-y));

  const float* simrow = (const float*)(ws + WOFF_SIM) + ((size_t)idx) * NPT;
  float best = -1e30f; int bi = 0;
  for (int j = 0; j < NPT; ++j) {
    float v = simrow[j];
    if (v > best) { best = v; bi = j; }
  }
  ((int*)(ws + WOFF_CORR))[idx] = bi;
}

// ---------------------------------------------------------------------------
// torch-style lower median mask + normalize (deterministic LDS reductions)
// ---------------------------------------------------------------------------
__global__ void median_kernel(char* ws) {
  __shared__ float sw[NPT];
  __shared__ float smed;
  __shared__ float sred[256];
  int b = blockIdx.x, t = threadIdx.x;
  const float* wv = (const float*)(ws + WOFF_WVEC) + b * NPT;
  sw[t] = wv[t]; sw[t + 256] = wv[t + 256];
  __syncthreads();
  #pragma unroll
  for (int e = 0; e < 2; ++e) {
    int i = t + e * 256;
    float v = sw[i];
    int cl = 0, ce = 0;
    for (int j = 0; j < NPT; ++j) { cl += (sw[j] < v); ce += (sw[j] == v); }
    if (cl <= 255 && 255 < cl + ce) smed = v;     // all writers agree on value
  }
  __syncthreads();
  float med = smed;
  float w0 = sw[t]       * ((sw[t]       >= med) ? 1.f : 0.f);
  float w1 = sw[t + 256] * ((sw[t + 256] >= med) ? 1.f : 0.f);
  sred[t] = w0 + w1;
  __syncthreads();
  for (int s = 128; s > 0; s >>= 1) {
    if (t < s) sred[t] += sred[t + s];
    __syncthreads();
  }
  float inv = 1.f / (sred[0] + 1e-8f);
  float* wn = (float*)(ws + WOFF_WNORM) + b * NPT;
  wn[t] = w0 * inv; wn[t + 256] = w1 * inv;
}

// ---------------------------------------------------------------------------
// weighted Kabsch (3x3 SVD via Jacobi) + rigid update, per batch
// ---------------------------------------------------------------------------
__device__ __forceinline__ float det3(const float M[3][3]) {
  return M[0][0] * (M[1][1] * M[2][2] - M[1][2] * M[2][1]) -
         M[0][1] * (M[1][0] * M[2][2] - M[1][2] * M[2][0]) +
         M[0][2] * (M[1][0] * M[2][1] - M[1][1] * M[2][0]);
}

__global__ void svd_kernel(PtrPack P, char* ws, float* out) {
  __shared__ float parts[64 * 9];
  __shared__ float red[15];
  __shared__ float sR[9], sT[3];
  const int b = blockIdx.x, t = threadIdx.x;
  float* srcC = (float*)(ws + WOFF_SRC);
  const float* wn = (const float*)(ws + WOFF_WNORM) + b * NPT;
  const int* corr = (const int*)(ws + WOFF_CORR) + b * NPT;
  const float* tgt = P.in[I_TGT];

  // pass 1: weighted means
  float ps[6] = {0, 0, 0, 0, 0, 0};
  for (int n = t; n < NPT; n += 64) {
    float w = wn[n];
    int j = corr[n];
    #pragma unroll
    for (int k = 0; k < 3; ++k) {
      ps[k]     += w * srcC[(b * 3 + k) * NPT + n];
      ps[3 + k] += w * tgt[(b * 3 + k) * NPT + j];
    }
  }
  #pragma unroll
  for (int k = 0; k < 6; ++k) parts[t * 6 + k] = ps[k];
  __syncthreads();
  if (t < 6) {
    float a = 0.f;
    for (int i = 0; i < 64; ++i) a += parts[i * 6 + t];
    red[t] = a;
  }
  __syncthreads();
  float srcm[3] = {red[0], red[1], red[2]}, tgtm[3] = {red[3], red[4], red[5]};
  // pass 2: H = sum w (s - srcm)(t - tgtm)^T
  float pH[9] = {0};
  for (int n = t; n < NPT; n += 64) {
    float w = wn[n];
    int j = corr[n];
    float sc[3], tc[3];
    #pragma unroll
    for (int k = 0; k < 3; ++k) {
      sc[k] = srcC[(b * 3 + k) * NPT + n] - srcm[k];
      tc[k] = tgt[(b * 3 + k) * NPT + j] - tgtm[k];
    }
    #pragma unroll
    for (int i = 0; i < 3; ++i)
      #pragma unroll
      for (int jj = 0; jj < 3; ++jj) pH[i * 3 + jj] += w * sc[i] * tc[jj];
  }
  __syncthreads();
  #pragma unroll
  for (int k = 0; k < 9; ++k) parts[t * 9 + k] = pH[k];
  __syncthreads();
  if (t < 9) {
    float a = 0.f;
    for (int i = 0; i < 64; ++i) a += parts[i * 9 + t];
    red[6 + t] = a;
  }
  __syncthreads();

  if (t == 0) {
    float H[3][3], A[3][3], V[3][3], Rm[3][3];
    for (int i = 0; i < 3; ++i)
      for (int j = 0; j < 3; ++j) H[i][j] = red[6 + i * 3 + j];
    for (int i = 0; i < 3; ++i)
      for (int j = 0; j < 3; ++j) {
        A[i][j] = H[0][i] * H[0][j] + H[1][i] * H[1][j] + H[2][i] * H[2][j];
        V[i][j] = (i == j) ? 1.f : 0.f;
      }
    for (int sweep = 0; sweep < 12; ++sweep)
      for (int pp = 0; pp < 3; ++pp) {
        int p = (pp == 2) ? 1 : 0, q = (pp == 0) ? 1 : 2;
        float apq = A[p][q];
        if (fabsf(apq) > 1e-14f) {
          float th = (A[q][q] - A[p][p]) / (2.f * apq);
          float tt = ((th >= 0.f) ? 1.f : -1.f) /
                     (fabsf(th) + sqrtf(th * th + 1.f));
          float cs = rsqrtf(tt * tt + 1.f), sn = tt * cs;
          for (int k = 0; k < 3; ++k) {
            float akp = A[k][p], akq = A[k][q];
            A[k][p] = cs * akp - sn * akq; A[k][q] = sn * akp + cs * akq;
          }
          for (int k = 0; k < 3; ++k) {
            float apk = A[p][k], aqk = A[q][k];
            A[p][k] = cs * apk - sn * aqk; A[q][k] = sn * apk + cs * aqk;
          }
          for (int k = 0; k < 3; ++k) {
            float vkp = V[k][p], vkq = V[k][q];
            V[k][p] = cs * vkp - sn * vkq; V[k][q] = sn * vkp + cs * vkq;
          }
        }
      }
    float e[3] = {A[0][0], A[1][1], A[2][2]};
    for (int a1 = 0; a1 < 2; ++a1)
      for (int b1 = a1 + 1; b1 < 3; ++b1)
        if (e[b1] > e[a1]) {
          float tmp = e[a1]; e[a1] = e[b1]; e[b1] = tmp;
          for (int k = 0; k < 3; ++k) {
            float tv = V[k][a1]; V[k][a1] = V[k][b1]; V[k][b1] = tv;
          }
        }
    float u0[3], u1[3], u2[3];
    for (int k = 0; k < 3; ++k)
      u0[k] = H[k][0] * V[0][0] + H[k][1] * V[1][0] + H[k][2] * V[2][0];
    float n0 = sqrtf(u0[0] * u0[0] + u0[1] * u0[1] + u0[2] * u0[2]);
    if (n0 < 1e-20f) {
      for (int i = 0; i < 3; ++i)
        for (int j = 0; j < 3; ++j) Rm[i][j] = (i == j) ? 1.f : 0.f;
    } else {
      for (int k = 0; k < 3; ++k) u0[k] /= n0;
      for (int k = 0; k < 3; ++k)
        u1[k] = H[k][0] * V[0][1] + H[k][1] * V[1][1] + H[k][2] * V[2][1];
      float dp = u1[0] * u0[0] + u1[1] * u0[1] + u1[2] * u0[2];
      for (int k = 0; k < 3; ++k) u1[k] -= dp * u0[k];
      float n1 = sqrtf(u1[0] * u1[0] + u1[1] * u1[1] + u1[2] * u1[2]);
      if (n1 < 1e-20f) {
        float ax[3] = {1.f, 0.f, 0.f};
        if (fabsf(u0[0]) > 0.9f) { ax[0] = 0.f; ax[1] = 1.f; }
        u1[0] = u0[1] * ax[2] - u0[2] * ax[1];
        u1[1] = u0[2] * ax[0] - u0[0] * ax[2];
        u1[2] = u0[0] * ax[1] - u0[1] * ax[0];
        n1 = sqrtf(u1[0] * u1[0] + u1[1] * u1[1] + u1[2] * u1[2]) + 1e-20f;
      }
      for (int k = 0; k < 3; ++k) u1[k] /= n1;
      u2[0] = u0[1] * u1[2] - u0[2] * u1[1];
      u2[1] = u0[2] * u1[0] - u0[0] * u1[2];
      u2[2] = u0[0] * u1[1] - u0[1] * u1[0];
      float R0[3][3];
      for (int i = 0; i < 3; ++i)
        for (int j = 0; j < 3; ++j)
          R0[i][j] = V[i][0] * u0[j] + V[i][1] * u1[j] + V[i][2] * u2[j];
      if (det3(R0) < 0.f)
        for (int i = 0; i < 3; ++i) V[i][2] = -V[i][2];
      for (int i = 0; i < 3; ++i)
        for (int j = 0; j < 3; ++j)
          Rm[i][j] = V[i][0] * u0[j] + V[i][1] * u1[j] + V[i][2] * u2[j];
    }
    float tab[3];
    for (int i = 0; i < 3; ++i)
      tab[i] = tgtm[i] -
               (Rm[i][0] * srcm[0] + Rm[i][1] * srcm[1] + Rm[i][2] * srcm[2]);
    // accumulate R, t and publish
    float* RT = (float*)(ws + WOFF_RACC);
    float Ro[3][3], to[3], Rn[3][3], tn[3];
    for (int i = 0; i < 3; ++i) {
      to[i] = RT[18 + b * 3 + i];
      for (int j = 0; j < 3; ++j) Ro[i][j] = RT[b * 9 + i * 3 + j];
    }
    for (int i = 0; i < 3; ++i) {
      tn[i] = Rm[i][0] * to[0] + Rm[i][1] * to[1] + Rm[i][2] * to[2] + tab[i];
      for (int j = 0; j < 3; ++j)
        Rn[i][j] = Rm[i][0] * Ro[0][j] + Rm[i][1] * Ro[1][j] + Rm[i][2] * Ro[2][j];
    }
    for (int i = 0; i < 3; ++i) {
      RT[18 + b * 3 + i] = tn[i];
      out[18 + b * 3 + i] = tn[i];
      sT[i] = tab[i];
      for (int j = 0; j < 3; ++j) {
        RT[b * 9 + i * 3 + j] = Rn[i][j];
        out[b * 9 + i * 3 + j] = Rn[i][j];
        sR[i * 3 + j] = Rm[i][j];
      }
    }
  }
  __syncthreads();
  // warp src forward for next iteration
  for (int n = t; n < NPT; n += 64) {
    float p0 = srcC[(b * 3 + 0) * NPT + n];
    float p1 = srcC[(b * 3 + 1) * NPT + n];
    float p2 = srcC[(b * 3 + 2) * NPT + n];
    float q0 = sR[0] * p0 + sR[1] * p1 + sR[2] * p2 + sT[0];
    float q1 = sR[3] * p0 + sR[4] * p1 + sR[5] * p2 + sT[1];
    float q2 = sR[6] * p0 + sR[7] * p1 + sR[8] * p2 + sT[2];
    srcC[(b * 3 + 0) * NPT + n] = q0;
    srcC[(b * 3 + 1) * NPT + n] = q1;
    srcC[(b * 3 + 2) * NPT + n] = q2;
  }
}

// ---------------------------------------------------------------------------
extern "C" void kernel_launch(void* const* d_in, const int* in_sizes, int n_in,
                              void* d_out, int out_size, void* d_ws,
                              size_t ws_size, hipStream_t stream) {
  (void)in_sizes; (void)out_size; (void)ws_size;
  PtrPack P;
  for (int i = 0; i < 80; ++i)
    P.in[i] = (i < n_in) ? (const float*)d_in[i] : nullptr;
  char* ws = (char*)d_ws;
  float* out = (float*)d_out;

  prep_kernel<<<1, 256, 0, stream>>>(P, ws);
  for (int it = 0; it < 2; ++it) {
    zero_wmax<<<64, 256, 0, stream>>>(ws);
    pair_kernel<<<4096, 256, 0, stream>>>(P, ws, it);
    weights_kernel<<<4, 256, 0, stream>>>(P, ws, it);
    median_kernel<<<NB, 256, 0, stream>>>(ws);
    svd_kernel<<<NB, 64, 0, stream>>>(P, ws, out);
  }
}
